// GraphConvolution_wmask_learnadj_65627100283120
// MI455X (gfx1250) — matrix operand order repaired and checked by reading, not compiled, not run on hardware
//
#include <hip/hip_runtime.h>

// ---------- types ----------
typedef __attribute__((ext_vector_type(16))) __bf16 v16bf;
typedef __attribute__((ext_vector_type(8)))  float  v8f;
typedef __attribute__((ext_vector_type(4)))  unsigned int uint4v;
typedef __attribute__((ext_vector_type(2)))  unsigned int uint2v;
typedef __attribute__((ext_vector_type(4)))  float  float4v;

struct Frag { union { v16bf v; uint4v q[2]; }; };

// ---------- constants ----------
#define BB 8
#define NN 4096
#define FF 512
#define OO 512
#define DD 64
#define CHUNK 1024
#define PSTR (CHUNK + 8)   // +16B row pad -> conflict-free ds_load_b128

// ---------- helpers ----------
__device__ __forceinline__ unsigned short f2bf(float f) {
    unsigned u = __float_as_uint(f);
    u += 0x7FFFu + ((u >> 16) & 1u);            // RNE
    return (unsigned short)(u >> 16);
}
__device__ __forceinline__ float bf2f(unsigned short h) {
    return __uint_as_float((unsigned)h << 16);
}
__device__ __forceinline__ unsigned pack2(float a, float b) {
    return (unsigned)f2bf(a) | ((unsigned)f2bf(b) << 16);
}
__device__ __forceinline__ v8f wmma_bf16(const Frag& a, const Frag& b, v8f c) {
    return __builtin_amdgcn_wmma_f32_16x16x32_bf16(false, a.v, false, b.v,
                                                   (short)0, c, false, false);
}

// ---------- fp32 -> bf16 bulk convert (x) ----------
__global__ __launch_bounds__(256) void k_cvt_x(const float* __restrict__ x,
                                               unsigned short* __restrict__ xb, int n4) {
    int i = blockIdx.x * blockDim.x + threadIdx.x;
    if (i >= n4) return;
    float4v f = ((const float4v*)x)[i];
    uint2v p; p[0] = pack2(f[0], f[1]); p[1] = pack2(f[2], f[3]);
    ((uint2v*)xb)[i] = p;
}

// ---------- weight transpose + convert: wT[dout][f] = bf16(w[f][dout]) ----------
__global__ __launch_bounds__(256) void k_cvt_wT(const float* __restrict__ w,
                                                unsigned short* __restrict__ wT,
                                                int Fdim, int Dout) {
    int i = blockIdx.x * blockDim.x + threadIdx.x;
    if (i >= Fdim * Dout) return;
    int dout = i / Fdim, f = i - dout * Fdim;
    wT[i] = f2bf(w[f * Dout + dout]);
}

// ---------- projection GEMM: out = xb[B*N,512] @ wT' + bias (bf16 WMMA) ----------
// TRANS=false: out row-major bf16 [B*N, Dout]     (q, k : Dout=64)
// TRANS=true : out supportT bf16 [B][Dout][4096]  (Dout=512, b128 stores)
template <bool TRANS>
__global__ __launch_bounds__(128) void k_proj(const unsigned short* __restrict__ xb,
                                              const unsigned short* __restrict__ wT,
                                              const float* __restrict__ bias,
                                              unsigned short* __restrict__ out, int Dout) {
    const int lane = threadIdx.x & 31, wv = threadIdx.x >> 5;
    const int l15 = lane & 15, hi = lane >> 4, mhalf = hi * 8;
    const int rt = blockIdx.x;                      // 16-row tile over B*N
    const int colbase = (blockIdx.y * 4 + wv) * 16; // 16-col tile
    const unsigned short* arow = xb + ((long)rt * 16 + l15) * FF;
    const unsigned short* brow = wT + (long)(colbase + l15) * FF;
    const int aoff = hi * 8, boff = hi * 8;

    v8f acc; for (int i = 0; i < 8; ++i) acc[i] = 0.f;
    #pragma unroll 4
    for (int k0 = 0; k0 < FF; k0 += 32) {
        Frag a, b;
        a.q[0] = *(const uint4v*)(arow + k0 + aoff);
        a.q[1] = *(const uint4v*)(arow + k0 + aoff + 16);
        b.q[0] = *(const uint4v*)(brow + k0 + boff);
        b.q[1] = *(const uint4v*)(brow + k0 + boff + 16);
        acc = wmma_bf16(a, b, acc);
    }
    if (TRANS) {
        long g0 = (long)rt * 16;                    // tile never straddles a batch
        int bt = (int)(g0 >> 12), nloc = (int)(g0 & (NN - 1));
        unsigned short* dst = out + ((long)bt * Dout + colbase + l15) * NN + nloc + mhalf;
        uint4v pk;
        pk[0] = pack2(acc[0], acc[1]); pk[1] = pack2(acc[2], acc[3]);
        pk[2] = pack2(acc[4], acc[5]); pk[3] = pack2(acc[6], acc[7]);
        *(uint4v*)dst = pk;                         // transposed store = contiguous b128
    } else {
        float bv = bias ? bias[colbase + l15] : 0.f;
        #pragma unroll
        for (int r = 0; r < 8; ++r)
            out[((long)rt * 16 + mhalf + r) * Dout + colbase + l15] = f2bf(acc[r] + bv);
    }
}

// ---------- fused attention: out = (mask/sum) * exp(q@k^T/8) @ support ----------
// 1 workgroup (16 waves) per (batch, 16 query rows). P chunk lives in LDS.
__global__ __launch_bounds__(512) void k_attn(const unsigned short* __restrict__ qb,
                                              const unsigned short* __restrict__ kb,
                                              const unsigned short* __restrict__ supT,
                                              const float* __restrict__ mask,
                                              float* __restrict__ out) {
    __shared__ __align__(16) unsigned short P[16][PSTR];
    __shared__ float rowsum[16];
    __shared__ float rowscale[16];

    const int lane = threadIdx.x & 31, wv = threadIdx.x >> 5;
    const int l15 = lane & 15, hi = lane >> 4, mhalf = hi * 8;
    const int bat = blockIdx.x >> 8;                // 256 q-tiles per batch
    const int m0  = (blockIdx.x & 255) * 16;
    const int aoff = hi * 8, boff = hi * 8;

    // q A-fragments (16x64) resident all kernel
    const unsigned short* qrow = qb + ((long)(bat * NN + m0 + l15)) * DD;
    Frag aq0, aq1;
    aq0.q[0] = *(const uint4v*)(qrow + aoff);
    aq0.q[1] = *(const uint4v*)(qrow + aoff + 16);
    aq1.q[0] = *(const uint4v*)(qrow + 32 + aoff);
    aq1.q[1] = *(const uint4v*)(qrow + 32 + aoff + 16);

    if (threadIdx.x < 16) rowsum[threadIdx.x] = 0.f;
    v8f acc0, acc1;
    for (int i = 0; i < 8; ++i) { acc0[i] = 0.f; acc1[i] = 0.f; }
    const int cb = wv * 32;                          // this wave's 32 output cols
    const unsigned short* bs0 = supT + ((long)(bat * OO + cb + l15)) * NN + boff;
    const unsigned short* bs1 = bs0 + (long)16 * NN;
    __syncthreads();

    for (int c = 0; c < NN / CHUNK; ++c) {
        const int kc0 = c * CHUNK;
        // ---- stage 1: P[16][CHUNK] = exp(sim/8), 4 key-tiles per wave ----
        #pragma unroll
        for (int t = 0; t < 4; ++t) {
            const int tt = wv * 4 + t;
            const unsigned short* krow = kb + ((long)(bat * NN + kc0 + tt * 16 + l15)) * DD;
            Frag bk;
            v8f s; for (int i = 0; i < 8; ++i) s[i] = 0.f;
            bk.q[0] = *(const uint4v*)(krow + boff);
            bk.q[1] = *(const uint4v*)(krow + boff + 16);
            s = wmma_bf16(aq0, bk, s);
            bk.q[0] = *(const uint4v*)(krow + 32 + boff);
            bk.q[1] = *(const uint4v*)(krow + 32 + boff + 16);
            s = wmma_bf16(aq1, bk, s);
            const int col = tt * 16 + l15;
            #pragma unroll
            for (int r = 0; r < 8; ++r)
                P[mhalf + r][col] = f2bf(__expf(s[r] * 0.125f));
        }
        __syncthreads();
        // ---- row sums: wave w reduces row w ----
        {
            float s = 0.f;
            for (int i = lane; i < CHUNK; i += 32) s += bf2f(P[wv][i]);
            for (int o = 16; o > 0; o >>= 1) s += __shfl_down(s, o, 32);
            if (lane == 0) rowsum[wv] += s;
        }
        // ---- stage 2: acc += P(16xCHUNK) @ supportT' (K=32 steps) ----
        for (int ks = 0; ks < CHUNK; ks += 32) {
            Frag a, b0, b1;
            a.q[0] = *(const uint4v*)&P[l15][ks + aoff];
            a.q[1] = *(const uint4v*)&P[l15][ks + aoff + 16];
            const unsigned short* p0 = bs0 + kc0 + ks;
            const unsigned short* p1 = bs1 + kc0 + ks;
            b0.q[0] = *(const uint4v*)p0; b0.q[1] = *(const uint4v*)(p0 + 16);
            b1.q[0] = *(const uint4v*)p1; b1.q[1] = *(const uint4v*)(p1 + 16);
            acc0 = wmma_bf16(a, b0, acc0);
            acc1 = wmma_bf16(a, b1, acc1);
        }
        __syncthreads();
    }
    // ---- finalize: scale by mask/rowsum, store fp32 ----
    if (threadIdx.x < 16)
        rowscale[threadIdx.x] = mask[bat * NN + m0 + threadIdx.x] / rowsum[threadIdx.x];
    __syncthreads();
    float* orow = out + ((long)(bat * NN + m0 + mhalf)) * OO + cb + l15;
    #pragma unroll
    for (int r = 0; r < 8; ++r) {
        float sc = rowscale[mhalf + r];
        orow[(long)r * OO]      = acc0[r] * sc;
        orow[(long)r * OO + 16] = acc1[r] * sc;
    }
}

// ---------- launcher ----------
extern "C" void kernel_launch(void* const* d_in, const int* in_sizes, int n_in,
                              void* d_out, int out_size, void* d_ws, size_t ws_size,
                              hipStream_t stream) {
    const float* x    = (const float*)d_in[0];
    const float* mask = (const float*)d_in[1];
    const float* w1   = (const float*)d_in[2];
    const float* b1   = (const float*)d_in[3];
    const float* w2   = (const float*)d_in[4];
    const float* b2   = (const float*)d_in[5];
    const float* wgt  = (const float*)d_in[6];
    float* out = (float*)d_out;

    // workspace layout (bf16 buffers)
    const size_t SZ_XB  = (size_t)BB * NN * FF * 2;   // 32 MiB
    const size_t SZ_WST = (size_t)DD * FF * 2;        // 64 KiB (w1T/w2T)
    const size_t SZ_WT  = (size_t)FF * OO * 2;        // 512 KiB
    const size_t SZ_QB  = (size_t)BB * NN * DD * 2;   // 4 MiB (qb/kb)
    char* base = (char*)d_ws;
    unsigned short* xb   = (unsigned short*)base;               base += SZ_XB;
    unsigned short* w1T  = (unsigned short*)base;               base += SZ_WST;
    unsigned short* w2T  = (unsigned short*)base;               base += SZ_WST;
    unsigned short* wT   = (unsigned short*)base;               base += SZ_WT;
    unsigned short* qb   = (unsigned short*)base;               base += SZ_QB;
    unsigned short* kb   = (unsigned short*)base;               base += SZ_QB;
    unsigned short* supT = (unsigned short*)base;

    // 1) converts
    int n4 = BB * NN * FF / 4;
    k_cvt_x<<<(n4 + 255) / 256, 256, 0, stream>>>(x, xb, n4);
    k_cvt_wT<<<(FF * DD + 255) / 256, 256, 0, stream>>>(w1, w1T, FF, DD);
    k_cvt_wT<<<(FF * DD + 255) / 256, 256, 0, stream>>>(w2, w2T, FF, DD);
    k_cvt_wT<<<(FF * OO + 255) / 256, 256, 0, stream>>>(wgt, wT, FF, OO);

    // 2) projections (WMMA bf16)
    const int rowTiles = BB * NN / 16;  // 2048
    k_proj<false><<<dim3(rowTiles, 1), 128, 0, stream>>>(xb, w1T, b1, qb, DD);
    k_proj<false><<<dim3(rowTiles, 1), 128, 0, stream>>>(xb, w2T, b2, kb, DD);
    k_proj<true ><<<dim3(rowTiles, 8), 128, 0, stream>>>(xb, wT, nullptr, supT, OO);

    // 3) fused masked-softmax attention (WMMA bf16, LDS-resident P)
    k_attn<<<rowTiles, 512, 0, stream>>>(qb, kb, supT, mask, out);
}
